// MultiHeadAttention_21904333209639
// MI455X (gfx1250) — matrix operand order repaired and checked by reading, not compiled, not run on hardware
//
#include <hip/hip_runtime.h>
#include <cstdint>

#define SB 2
#define SH 16
#define SS 2048
#define SD 1024
#define DH 64
#define NREL 257

typedef __attribute__((ext_vector_type(16))) __bf16 bf16x16;
typedef __attribute__((ext_vector_type(8)))  float  f32x8;

union Frag {
  bf16x16 v;
  unsigned u[8];
  unsigned short h[16];
};

__device__ inline unsigned short f2bf(float f) {
  unsigned u = __float_as_uint(f);
  unsigned r = (u + 0x7FFFu + ((u >> 16) & 1u)) >> 16;
  return (unsigned short)r;
}
__device__ inline float bf2f(unsigned short h) {
  return __uint_as_float(((unsigned)h) << 16);
}

// LDS byte offset of a __shared__ object (generic-pointer low 32 bits are the
// LDS aperture offset on gfx1250).
__device__ inline unsigned lds_off(const void* p) {
  return (unsigned)(size_t)p;
}

// gfx1250 async global->LDS copy, 16B per lane, tracked by ASYNCcnt.
__device__ inline void async_g2l_b128(unsigned lds_addr, const void* gptr) {
  uint64_t ga = (uint64_t)gptr;
  asm volatile("global_load_async_to_lds_b128 %0, %1, off"
               :: "v"(lds_addr), "v"(ga) : "memory");
}
__device__ inline void wait_async0() {
#if __has_builtin(__builtin_amdgcn_s_wait_asynccnt)
  __builtin_amdgcn_s_wait_asynccnt(0);
#else
  asm volatile("s_wait_asynccnt 0" ::: "memory");
#endif
}

// ---------------------------------------------------------------------------
// Generic GEMM: C[M,N] = A[M,K] * W[K,N] + bias, bf16 WMMA, f32 accumulate.
// A_BF16: 0 -> A f32 (convert while staging, prefetch next tile),
//         1 -> A bf16 (raw async copy to LDS).
// OUT_MODE: 0 -> bf16 head-split [B,H,S,64];  1 -> f32 flat [M,N].
// Block tile 128(M) x 64(N); 8 waves as 4x2, each wave 32x32 (2x2 WMMA).
// ---------------------------------------------------------------------------
template <int A_BF16, int OUT_MODE>
__global__ __launch_bounds__(256)
void gemm_bf16_kernel(const void* __restrict__ Aptr,
                      const float* __restrict__ W, const float* __restrict__ bias,
                      void* __restrict__ Out,
                      int M, int N, int K)
{
  __shared__ __align__(16) unsigned short As[128][32];
  __shared__ __align__(16) unsigned short Bs[64][32];

  const int tid  = threadIdx.x;
  const int lane = tid & 31;
  const int w    = tid >> 5;
  const int wm   = w & 3;        // 4 M-slots of 32 rows
  const int wn   = w >> 2;       // 2 N-slots of 32 cols
  const int row0 = blockIdx.y * 128;
  const int col0 = blockIdx.x * 64;
  const int half = lane >> 4;
  const int l16  = lane & 15;

  const f32x8 z = {0.f,0.f,0.f,0.f,0.f,0.f,0.f,0.f};
  f32x8 acc[2][2];
  acc[0][0] = z; acc[0][1] = z; acc[1][0] = z; acc[1][1] = z;

  const int am = tid >> 1;            // A staging: half row per thread
  const int akb = (tid & 1) * 16;
  const int bk = tid >> 3;            // B staging
  const int bn0 = (tid & 7) * 8;

  for (int kk = 0; kk < K; kk += 32) {
    // ---- stage A tile 128x32 ----
    if (A_BF16) {
      unsigned lds = lds_off(&As[am][akb]);
      const unsigned short* g =
          (const unsigned short*)Aptr + (size_t)(row0 + am) * K + kk + akb;
      async_g2l_b128(lds, g);
      async_g2l_b128(lds + 16, g + 8);
    } else {
      const float* srcf = (const float*)Aptr + (size_t)(row0 + am) * K + kk + akb;
      if (kk + 32 < K) __builtin_prefetch(srcf + 32, 0, 1);
      const float4* src = (const float4*)srcf;
      #pragma unroll
      for (int j = 0; j < 4; ++j) {
        float4 f = src[j];
        As[am][akb + 4*j + 0] = f2bf(f.x);
        As[am][akb + 4*j + 1] = f2bf(f.y);
        As[am][akb + 4*j + 2] = f2bf(f.z);
        As[am][akb + 4*j + 3] = f2bf(f.w);
      }
    }
    // ---- stage B tile 32x64 transposed into Bs[n][k] ----
    {
      const float* srcf = W + (size_t)(kk + bk) * N + col0 + bn0;
      if (kk + 32 < K) __builtin_prefetch(srcf + (size_t)32 * N, 0, 1);
      const float4* src = (const float4*)srcf;
      #pragma unroll
      for (int j = 0; j < 2; ++j) {
        float4 f = src[j];
        Bs[bn0 + 4*j + 0][bk] = f2bf(f.x);
        Bs[bn0 + 4*j + 1][bk] = f2bf(f.y);
        Bs[bn0 + 4*j + 2][bk] = f2bf(f.z);
        Bs[bn0 + 4*j + 3][bk] = f2bf(f.w);
      }
    }
    if (A_BF16) wait_async0();
    __syncthreads();

    Frag a[2], b[2];
    #pragma unroll
    for (int mi = 0; mi < 2; ++mi) {
      const unsigned* r = (const unsigned*)&As[wm*32 + mi*16 + l16][0];
      #pragma unroll
      for (int v = 0; v < 8; ++v) {
        int idx = (v < 4) ? (v + 4*half) : (8 + (v - 4) + 4*half);
        a[mi].u[v] = r[idx];
      }
    }
    #pragma unroll
    for (int ni = 0; ni < 2; ++ni) {
      const unsigned* r = (const unsigned*)&Bs[wn*32 + ni*16 + l16][0];
      #pragma unroll
      for (int v = 0; v < 8; ++v) b[ni].u[v] = r[(v < 4) ? (v + 4*half) : (8 + (v - 4) + 4*half)];
    }
    #pragma unroll
    for (int mi = 0; mi < 2; ++mi)
      #pragma unroll
      for (int ni = 0; ni < 2; ++ni)
        acc[mi][ni] = __builtin_amdgcn_wmma_f32_16x16x32_bf16(
            false, a[mi].v, false, b[ni].v, (short)0, acc[mi][ni], false, false);
    __syncthreads();
  }

  // ---- epilogue ----
  #pragma unroll
  for (int mi = 0; mi < 2; ++mi) {
    #pragma unroll
    for (int ni = 0; ni < 2; ++ni) {
      int ncol = col0 + wn*32 + ni*16 + l16;
      float bv = bias[ncol];
      #pragma unroll
      for (int r = 0; r < 8; ++r) {
        int mrow = row0 + wm*32 + mi*16 + r + 8*half;
        float val = acc[mi][ni][r] + bv;
        if (OUT_MODE == 0) {
          int bb = mrow / SS, s = mrow % SS;
          int h = ncol >> 6, d = ncol & 63;
          ((unsigned short*)Out)[(((size_t)(bb * SH + h)) * SS + s) * DH + d] = f2bf(val);
        } else {
          ((float*)Out)[(size_t)mrow * N + ncol] = val;
        }
      }
    }
  }
}

// ---------------------------------------------------------------------------
// Logits + relative-K + softmax. One 16-row q-tile per block, 8 waves cover
// 256 keys each (16 column tiles). Writes normalized attn and L/R tail sums.
// ---------------------------------------------------------------------------
__global__ __launch_bounds__(256)
void attn_softmax_kernel(const unsigned short* __restrict__ Qh,
                         const unsigned short* __restrict__ Kh,
                         const float* __restrict__ relk,
                         const float* __restrict__ mask,
                         float* __restrict__ attn,
                         float* __restrict__ LR)
{
  __shared__ __align__(16) unsigned short Qs[16][64];
  __shared__ float P[16][NREL + 3];
  __shared__ float pmax[8][16], psum[8][16], pL[8][16], pR[8][16];
  __shared__ float rmax[16], rsum[16];

  const int tid = threadIdx.x, lane = tid & 31, w = tid >> 5;
  const int half = lane >> 4, l16 = lane & 15;
  const int nqt = SS / 16;
  const int bh = blockIdx.x / nqt;
  const int q0 = (blockIdx.x % nqt) * 16;
  const int bb = bh / SH;

  const unsigned short* Qbase = Qh + ((size_t)bh * SS + q0) * DH;
  for (int e = tid; e < 16 * 64; e += 256)
    ((unsigned short*)Qs)[e] = Qbase[(size_t)(e >> 6) * DH + (e & 63)];
  __syncthreads();

  // rel-K projection table P[row][j] = q_row . rel_k_emb[j]
  for (int idx = tid; idx < 16 * NREL; idx += 256) {
    int row = idx / NREL, j = idx % NREL;
    const float* rp = relk + (size_t)j * DH;
    float s = 0.f;
    #pragma unroll 8
    for (int d = 0; d < DH; ++d) s += bf2f(Qs[row][d]) * rp[d];
    P[row][j] = s;
  }
  __syncthreads();

  // hoisted A fragments (two 32-deep k-steps covering depth 64)
  Frag a[2];
  {
    const unsigned* r = (const unsigned*)&Qs[l16][0];
    #pragma unroll
    for (int ks = 0; ks < 2; ++ks)
      #pragma unroll
      for (int v = 0; v < 8; ++v) {
        int idx = ((v < 4) ? (v + 4*half) : (8 + (v - 4) + 4*half)) + 16 * ks;
        a[ks].u[v] = r[idx];
      }
  }

  const f32x8 z = {0.f,0.f,0.f,0.f,0.f,0.f,0.f,0.f};
  f32x8 acc[16];
  const unsigned short* Kbase = Kh + (size_t)bh * SS * DH;
  #pragma unroll
  for (int ct = 0; ct < 16; ++ct) {
    int n0 = w * 256 + ct * 16;
    const unsigned* kr = (const unsigned*)(Kbase + (size_t)(n0 + l16) * DH);
    if (ct + 1 < 16)
      __builtin_prefetch(Kbase + (size_t)(n0 + 16 + l16) * DH, 0, 1);
    Frag b0, b1;
    #pragma unroll
    for (int v = 0; v < 8; ++v) { int bi = (v < 4) ? (v + 4*half) : (8 + (v - 4) + 4*half); b0.u[v] = kr[bi]; b1.u[v] = kr[bi + 16]; }
    f32x8 c = z;
    c = __builtin_amdgcn_wmma_f32_16x16x32_bf16(false, a[0].v, false, b0.v, (short)0, c, false, false);
    c = __builtin_amdgcn_wmma_f32_16x16x32_bf16(false, a[1].v, false, b1.v, (short)0, c, false, false);
    acc[ct] = c;
  }

  // rel + scale + mask
  const float* mrow = mask + (size_t)bb * SS;
  #pragma unroll
  for (int ct = 0; ct < 16; ++ct) {
    int col = w * 256 + ct * 16 + l16;
    float mv = mrow[col] * -1e9f;
    #pragma unroll
    for (int r = 0; r < 8; ++r) {
      int lr = r + 8 * half;
      int dj = col - (q0 + lr);
      dj = dj < -128 ? -128 : (dj > 128 ? 128 : dj);
      acc[ct][r] = (acc[ct][r] + P[lr][dj + 128]) * 0.125f + mv;
    }
  }

  // row max
  #pragma unroll
  for (int r = 0; r < 8; ++r) {
    float m = -3.4e38f;
    #pragma unroll
    for (int ct = 0; ct < 16; ++ct) m = fmaxf(m, acc[ct][r]);
    #pragma unroll
    for (int off = 1; off < 16; off <<= 1) m = fmaxf(m, __shfl_xor(m, off, 32));
    if (l16 == 0) pmax[w][r + 8 * half] = m;
  }
  __syncthreads();
  if (tid < 16) {
    float m = pmax[0][tid];
    for (int ww = 1; ww < 8; ++ww) m = fmaxf(m, pmax[ww][tid]);
    rmax[tid] = m;
  }
  __syncthreads();

  // exp + sums (+ L/R tail buckets for the rel-V pass)
  #pragma unroll
  for (int r = 0; r < 8; ++r) {
    int lr = r + 8 * half;
    int q = q0 + lr;
    float rm = rmax[lr];
    float s = 0.f, sl = 0.f, sr = 0.f;
    #pragma unroll
    for (int ct = 0; ct < 16; ++ct) {
      int col = w * 256 + ct * 16 + l16;
      float e = __expf(acc[ct][r] - rm);
      acc[ct][r] = e;
      s += e;
      if (col <= q - 128) sl += e;
      else if (col >= q + 128) sr += e;
    }
    #pragma unroll
    for (int off = 1; off < 16; off <<= 1) {
      s  += __shfl_xor(s,  off, 32);
      sl += __shfl_xor(sl, off, 32);
      sr += __shfl_xor(sr, off, 32);
    }
    if (l16 == 0) { psum[w][lr] = s; pL[w][lr] = sl; pR[w][lr] = sr; }
  }
  __syncthreads();
  if (tid < 16) {
    float s = 0.f, sl = 0.f, sr = 0.f;
    for (int ww = 0; ww < 8; ++ww) { s += psum[ww][tid]; sl += pL[ww][tid]; sr += pR[ww][tid]; }
    rsum[tid] = s;
    float inv = 1.f / s;
    size_t lri = ((size_t)bh * SS + q0 + tid) * 2;
    LR[lri] = sl * inv; LR[lri + 1] = sr * inv;
  }
  __syncthreads();

  // write normalized attention
  float* arow = attn + (size_t)bh * SS * SS;
  #pragma unroll
  for (int ct = 0; ct < 16; ++ct) {
    int col = w * 256 + ct * 16 + l16;
    #pragma unroll
    for (int r = 0; r < 8; ++r) {
      int lr = r + 8 * half;
      arow[(size_t)(q0 + lr) * SS + col] = acc[ct][r] / rsum[lr];
    }
  }
}

// ---------------------------------------------------------------------------
// attn @ V (WMMA over bf16, async-staged raw V tile in LDS) + rel-V band.
// Writes head-concatenated bf16 activations for the final projection.
// ---------------------------------------------------------------------------
__global__ __launch_bounds__(256)
void attn_v_kernel(const float* __restrict__ attn,
                   const unsigned short* __restrict__ Vh,
                   const float* __restrict__ relv,
                   const float* __restrict__ LR,
                   unsigned short* __restrict__ Cc)
{
  __shared__ __align__(16) char smem[32 * 1024]; // aliased: Vraw / Opart / band
  __shared__ float LRs[16][2];

  unsigned short* Vraw = (unsigned short*)smem;  // [256 keys][64 d] bf16 (row-major)
  float* Opart = (float*)smem;                   // [8][16][64] wave partials
  float* band  = (float*)smem;                   // [16][256] attn diag band

  const int tid = threadIdx.x, lane = tid & 31, w = tid >> 5;
  const int half = lane >> 4, l16 = lane & 15;
  const int nqt = SS / 16;
  const int bh = blockIdx.x / nqt;
  const int q0 = (blockIdx.x % nqt) * 16;
  const int bb = bh / SH, hh = bh % SH;

  const float* Arow = attn + (size_t)bh * SS * SS;
  const unsigned short* Vbase = Vh + (size_t)bh * SS * DH;

  const f32x8 z = {0.f,0.f,0.f,0.f,0.f,0.f,0.f,0.f};
  f32x8 acc[4];
  acc[0] = z; acc[1] = z; acc[2] = z; acc[3] = z;

  const float* aq = Arow + (size_t)(q0 + l16) * SS;
  const unsigned vlds = lds_off(&Vraw[(size_t)tid * DH]);

  for (int si = 0; si < 8; ++si) {
    // async-stage 256 keys of V (raw row-major copy, 128B per thread)
    {
      const unsigned short* g = Vbase + (size_t)(si * 256 + tid) * DH;
      #pragma unroll
      for (int j = 0; j < 8; ++j)
        async_g2l_b128(vlds + 16 * j, g + 8 * j);
      wait_async0();
    }
    __syncthreads();

    int kb = si * 256 + w * 32;
    Frag a;
    #pragma unroll
    for (int v = 0; v < 8; ++v) {
      int k = kb + ((v < 4) ? (2*v + 8*half) : (16 + 2*(v - 4) + 8*half));
      float2 f = *(const float2*)(aq + k);
      a.u[v] = (unsigned)f2bf(f.x) | ((unsigned)f2bf(f.y) << 16);
    }
    #pragma unroll
    for (int ni = 0; ni < 4; ++ni) {
      Frag b;
      #pragma unroll
      for (int v = 0; v < 8; ++v) {
        int k0 = w * 32 + ((v < 4) ? (2*v + 8*half) : (16 + 2*(v - 4) + 8*half));   // key pair within staged tile
        int d  = ni * 16 + l16;
        unsigned lo = Vraw[(size_t)k0 * DH + d];
        unsigned hi = Vraw[(size_t)(k0 + 1) * DH + d];
        b.u[v] = lo | (hi << 16);
      }
      acc[ni] = __builtin_amdgcn_wmma_f32_16x16x32_bf16(
          false, a.v, false, b.v, (short)0, acc[ni], false, false);
    }
    __syncthreads();
  }

  // cross-wave reduction via LDS
  #pragma unroll
  for (int ni = 0; ni < 4; ++ni)
    #pragma unroll
    for (int r = 0; r < 8; ++r)
      Opart[(w*16 + r + 8*half) * 64 + ni*16 + l16] = acc[ni][r];
  __syncthreads();

  float val[4]; int orow[4], od[4];
  #pragma unroll
  for (int i = 0; i < 4; ++i) {
    int oidx = tid * 4 + i;
    orow[i] = oidx >> 6; od[i] = oidx & 63;
    float s = 0.f;
    for (int ww = 0; ww < 8; ++ww) s += Opart[(ww*16 + orow[i]) * 64 + od[i]];
    val[i] = s;
  }
  __syncthreads();

  // stage the 255-wide attn diagonal band + tail buckets
  for (int e = tid; e < 16 * 255; e += 256) {
    int row = e / 255, i = e % 255;
    int key = q0 + row + i - 127;
    band[row * 256 + i] = (key >= 0 && key < SS)
        ? Arow[(size_t)(q0 + row) * SS + key] : 0.f;
  }
  if (tid < 32) {
    int row = tid >> 1, which = tid & 1;
    LRs[row][which] = LR[((size_t)bh * SS + q0 + row) * 2 + which];
  }
  __syncthreads();

  #pragma unroll
  for (int i = 0; i < 4; ++i) {
    int row = orow[i], d = od[i];
    float s = val[i] + LRs[row][0] * relv[d] + LRs[row][1] * relv[256 * DH + d];
    const float* brow = band + row * 256;
    #pragma unroll 4
    for (int t = 0; t < 255; ++t) s += brow[t] * relv[(t + 1) * DH + d];
    Cc[((size_t)(bb * SS + q0 + row)) * SD + hh * DH + d] = f2bf(s);
  }
}

// ---------------------------------------------------------------------------
extern "C" void kernel_launch(void* const* d_in, const int* in_sizes, int n_in,
                              void* d_out, int out_size, void* d_ws, size_t ws_size,
                              hipStream_t stream)
{
  (void)in_sizes; (void)n_in; (void)out_size; (void)ws_size;
  const float* q    = (const float*)d_in[0];
  const float* k    = (const float*)d_in[1];
  const float* v    = (const float*)d_in[2];
  const float* mask = (const float*)d_in[3];
  const float* wq   = (const float*)d_in[4];
  const float* bq   = (const float*)d_in[5];
  const float* wk   = (const float*)d_in[6];
  const float* bk   = (const float*)d_in[7];
  const float* wv   = (const float*)d_in[8];
  const float* bv   = (const float*)d_in[9];
  const float* wo   = (const float*)d_in[10];
  const float* bo   = (const float*)d_in[11];
  const float* relk = (const float*)d_in[12];
  const float* relv = (const float*)d_in[13];

  float* out  = (float*)d_out;
  float* attn = out + (size_t)SB * SS * SD;   // 4,194,304 floats offset

  char* ws = (char*)d_ws;
  unsigned short* Qh = (unsigned short*)(ws + 0);
  unsigned short* Kh = (unsigned short*)(ws + 8388608);
  unsigned short* Vh = (unsigned short*)(ws + 16777216);
  unsigned short* Cc = (unsigned short*)(ws + 25165824);
  float*          LR = (float*)(ws + 33554432);

  const int M = SB * SS; // 4096
  dim3 gGemm(SD / 64, M / 128);
  gemm_bf16_kernel<0,0><<<gGemm, 256, 0, stream>>>(q, wq, bq, Qh, M, SD, SD);
  gemm_bf16_kernel<0,0><<<gGemm, 256, 0, stream>>>(k, wk, bk, Kh, M, SD, SD);
  gemm_bf16_kernel<0,0><<<gGemm, 256, 0, stream>>>(v, wv, bv, Vh, M, SD, SD);

  dim3 gAttn(SB * SH * (SS / 16));
  attn_softmax_kernel<<<gAttn, 256, 0, stream>>>(Qh, Kh, relk, mask, attn, LR);
  attn_v_kernel<<<gAttn, 256, 0, stream>>>(attn, Vh, relv, LR, Cc);

  gemm_bf16_kernel<1,1><<<gGemm, 256, 0, stream>>>(Cc, wo, bo, out, M, SD, SD);
}
